// ConvLayer_455266533542
// MI455X (gfx1250) — matrix operand order repaired and checked by reading, not compiled, not run on hardware
//
#include <hip/hip_runtime.h>
#include <hip/hip_bf16.h>

typedef __attribute__((ext_vector_type(16))) __bf16 v16bf;
typedef __attribute__((ext_vector_type(8)))  __bf16 v8bf;
typedef __attribute__((ext_vector_type(8)))  float  v8f;
typedef unsigned int u32x4 __attribute__((ext_vector_type(4)));
typedef int          i32x8 __attribute__((ext_vector_type(8)));
typedef int          i32x4 __attribute__((ext_vector_type(4)));

#define NB    16
#define CIN   256
#define COUT  512
#define HS    128
#define HP    132
#define OUTHW 64

// xb: [16][132][132][256] bf16 zero-padded NHWC
static constexpr size_t XB_ELEMS  = (size_t)NB * HP * HP * CIN;      // 71,368,704
// bsw: [ktile=288][ntile=32][lane=32][e=16] bf16, WMMA B-fragment order
static constexpr size_t BSW_ELEMS = (size_t)288 * 32 * 32 * 16;      // 4,718,592

// ---------------------------------------------------------------------------
// Kernel 1: fp32 NCHW -> bf16 zero-padded NHWC via LDS transpose tile.
// ---------------------------------------------------------------------------
__global__ __launch_bounds__(256) void pad_convert(const float* __restrict__ x,
                                                   __bf16* __restrict__ xb)
{
    __shared__ __bf16 tile[256][32];
    const int t  = threadIdx.x;
    const int w0 = blockIdx.x * 32;
    const int h  = blockIdx.y;            // 0..131 (padded coords)
    const int b  = blockIdx.z;
    const int hs = h - 2;
    const int wl = t & 31;
    const int ws = w0 + wl - 2;
    const bool ok = (hs >= 0) & (hs < HS) & (ws >= 0) & (ws < HS);

    #pragma unroll
    for (int cc = 0; cc < 32; ++cc) {
        const int c = cc * 8 + (t >> 5);
        float v = 0.0f;
        if (ok) v = x[(((size_t)b * CIN + c) * HS + hs) * HS + ws];   // coalesced along ws
        tile[c][wl] = (__bf16)v;
    }
    __syncthreads();
    #pragma unroll 4
    for (int wi = 0; wi < 32; ++wi) {
        const int w = w0 + wi;
        if (w < HP)
            xb[(((size_t)b * HP + h) * HP + w) * CIN + t] = tile[t][wi]; // 512B contiguous
    }
}

// ---------------------------------------------------------------------------
// Kernel 2: fuse blur(4x4) into conv(3x3) -> 6x6 kernel E, scale 1/48,
// emit bf16 pre-swizzled in WMMA B-fragment order. k = tap*256+c, tap=a*6+b'.
// ---------------------------------------------------------------------------
__global__ __launch_bounds__(256) void weight_xform(const float* __restrict__ wsrc,
                                                    __bf16* __restrict__ bsw)
{
    const float kb[4] = {1.f, 3.f, 3.f, 1.f};
    const size_t idx = (size_t)blockIdx.x * 256 + threadIdx.x;       // < BSW_ELEMS
    const int e     = (int)(idx & 15);
    const int lane  = (int)(idx >> 4) & 31;
    const int nb    = (int)(idx >> 9) & 31;
    const int ktile = (int)(idx >> 14);                               // 0..287
    const int n = nb * 16 + (lane & 15);
    const int k = ktile * 32 + (e >> 3) * 16 + (lane >> 4) * 8 + (e & 7);
    const int tap = k >> 8, c = k & 255;
    const int a = tap / 6, bq = tap % 6;

    float acc = 0.0f;
    #pragma unroll
    for (int r = 0; r < 3; ++r) {
        const int p = a - r;
        if (p < 0 || p > 3) continue;
        #pragma unroll
        for (int s = 0; s < 3; ++s) {
            const int q = bq - s;
            if (q < 0 || q > 3) continue;
            acc += wsrc[(((size_t)(r * 3 + s)) * CIN + c) * COUT + n] * (kb[p] * kb[q]);
        }
    }
    acc *= (1.0f / 64.0f) * (1.0f / 48.0f);   // blur norm * 1/sqrt(2304)
    bsw[idx] = (__bf16)acc;
}

// ---------------------------------------------------------------------------
// TDM descriptor issue: 2D tile, rows of 4096 bf16 (8 n-tiles of one k-tile),
// 4 rows (k-tiles) at stride 16384 elems  =>  32KB chunk into LDS.
// D# per CDNA5 ISA 8.3/8.4; groups 2/3 zero (<=2D tensor).
// This toolchain declares the 6-arg builtin: (g0, g1, g2, g3, g4, cpol).
// ---------------------------------------------------------------------------
__device__ __forceinline__ void tdm_load_b(const __bf16* gptr, unsigned int lds_off)
{
    unsigned long long ga = (unsigned long long)(uintptr_t)gptr;
    u32x4 g0 = { 1u,                                   // count=1, no gather
                 lds_off,                              // lds_addr (bytes)
                 (unsigned int)ga,                     // global_addr[31:0]
                 (unsigned int)(ga >> 32) | (2u << 30) // global_addr[56:32] | type=2
               };
    i32x8 g1 = { (int)(1u << 16),       // wg_mask=0, data_size=1 (2 bytes)
                 (int)(0x4000u << 16),  // tensor_dim0 = 16384 (lo16 in [31:16])
                 0,                     // tensor_dim0 hi16=0, tensor_dim1 lo16=0
                 (int)(0x1000u << 16) | 4, // tensor_dim1 hi16=4 (dim1=1<<18), tile_dim0=4096
                 4,                     // tile_dim1=4, tile_dim2=0
                 16384,                 // tensor_dim0_stride lo32
                 0, 0 };                // stride hi, dim1_stride (unused)
    i32x4 gz4 = { 0, 0, 0, 0 };
    i32x8 gz8 = { 0, 0, 0, 0, 0, 0, 0, 0 };
    __builtin_amdgcn_tensor_load_to_lds(g0, g1, gz4, gz4, gz8, 0);
}

// ---------------------------------------------------------------------------
// Kernel 3: implicit GEMM, M=(b,u,v)=65536, N=512, K=9216, bf16 WMMA.
// Block 128 thr (4 waves): tile M=32 x N=128; wave tile 16 x 64 (4 accums).
// B staged by TDM into LDS, double-buffered 32KB chunks (4 k-tiles each).
// ---------------------------------------------------------------------------
__global__ __launch_bounds__(128) void conv_wmma(const __bf16* __restrict__ xb,
                                                 const __bf16* __restrict__ bsw,
                                                 const float* __restrict__ bias,
                                                 float* __restrict__ out)
{
    __shared__ __align__(128) char smem[65536];   // 2 x 32KB B buffers; epilogue aliases buf0

    const int tid  = threadIdx.x;
    const int wave = tid >> 5;
    const int lane = tid & 31;
    const int lm   = lane & 15;
    const int half = lane >> 4;
    const int wm = (wave & 1) * 16;           // wave m-offset in block
    const int wn = (wave >> 1) * 64;          // wave n-offset in block

    const int mblk = blockIdx.x * 32;
    const int nblk = blockIdx.y * 128;
    const int b  = mblk >> 12;
    const int u  = (mblk >> 6) & 63;
    const int v0 = mblk & 63;
    const int v  = v0 + wm + lm;

    // A: lane holds M=lm; elements 0..7 = K half*8+0..7, 8..15 = K 16+half*8+0..7
    const __bf16* aBase = xb + (((size_t)(b * HP + 2 * u)) * HP + 2 * v) * CIN + half * 8;
    const __bf16* bswWG = bsw + (size_t)(nblk >> 4) * 512;   // this block's 8 n-tiles

    v8f acc0 = {}, acc1 = {}, acc2 = {}, acc3 = {};

    const int NCHUNK = 72;                    // 288 k-tiles / 4 per chunk
    if (wave == 0)
        tdm_load_b(bswWG, (unsigned int)(uintptr_t)(smem));

    for (int c = 0; c < NCHUNK; ++c) {
        if (wave == 0) {
            if (c + 1 < NCHUNK) {
                tdm_load_b(bswWG + (size_t)(c + 1) * 4 * 16384,
                           (unsigned int)(uintptr_t)(smem + ((c + 1) & 1) * 32768));
                __builtin_amdgcn_s_wait_tensorcnt(1);   // chunk c landed
            } else {
                __builtin_amdgcn_s_wait_tensorcnt(0);
            }
        }
        __syncthreads();                       // chunk c visible to all waves

        const char* bufp = smem + (c & 1) * 32768;
        {   // prefetch next chunk's first A row
            const int kk = (c + 1 < NCHUNK ? (c + 1) : c) * 4;
            const int tap = kk >> 3, ta = (kk >> 3) / 6, tb = tap - ta * 6;
            __builtin_prefetch(aBase + (ta * HP + tb) * CIN + (kk & 7) * 32, 0, 1);
        }
        #pragma unroll
        for (int kt = 0; kt < 4; ++kt) {
            const int kk = c * 4 + kt;
            const int tap = kk >> 3, cb = kk & 7;
            const int ta = tap / 6, tb = tap - ta * 6;
            const __bf16* ap = aBase + (ta * HP + tb) * CIN + cb * 32;
            v8bf lo = *(const v8bf*)(ap);
            v8bf hi = *(const v8bf*)(ap + 16);
            v16bf afrag = __builtin_shufflevector(lo, hi,
                0, 1, 2, 3, 4, 5, 6, 7, 8, 9, 10, 11, 12, 13, 14, 15);
            // B fragments from LDS: [kt][ntile-local][lane][16], 32B/lane contiguous
            const char* bl = bufp + kt * 8192 + (wn >> 4) * 1024 + lane * 32;
            v16bf b0 = *(const v16bf*)(bl);
            v16bf b1 = *(const v16bf*)(bl + 1024);
            v16bf b2 = *(const v16bf*)(bl + 2048);
            v16bf b3 = *(const v16bf*)(bl + 3072);
            acc0 = __builtin_amdgcn_wmma_f32_16x16x32_bf16(false, afrag, false, b0, (short)0, acc0, false, false);
            acc1 = __builtin_amdgcn_wmma_f32_16x16x32_bf16(false, afrag, false, b1, (short)0, acc1, false, false);
            acc2 = __builtin_amdgcn_wmma_f32_16x16x32_bf16(false, afrag, false, b2, (short)0, acc2, false, false);
            acc3 = __builtin_amdgcn_wmma_f32_16x16x32_bf16(false, afrag, false, b3, (short)0, acc3, false, false);
        }
        __syncthreads();                       // done reading buf (c&1) before it is refilled
    }

    // Epilogue: bias + leaky_relu(0.2) * sqrt(2); LDS transpose (stride 17)
    // for coalesced 128-bit stores along the NCHW v dimension.
    float* wl = (float*)smem + wave * (64 * 17);   // aliases B buffer 0 (K-loop done)
    const float sq2 = 1.41421356237f;
    v8f accs[4] = {acc0, acc1, acc2, acc3};
    #pragma unroll
    for (int s = 0; s < 4; ++s) {
        const float bv = bias[nblk + wn + s * 16 + lm];   // C/D: lane holds N=lm
        #pragma unroll
        for (int r = 0; r < 8; ++r) {                     // C/D: vgpr r -> M=r+8*half
            float val = accs[s][r] + bv;
            val = (val > 0.0f ? val : val * 0.2f) * sq2;
            wl[(s * 16 + lm) * 17 + (r + 8 * half)] = val;
        }
    }
    __syncthreads();
    #pragma unroll
    for (int j = 0; j < 8; ++j) {
        const int nl = j * 8 + (lane >> 2);
        const int ml = (lane & 3) * 4;
        const float x0 = wl[nl * 17 + ml + 0];
        const float x1 = wl[nl * 17 + ml + 1];
        const float x2 = wl[nl * 17 + ml + 2];
        const float x3 = wl[nl * 17 + ml + 3];
        const size_t ng  = (size_t)nblk + wn + nl;
        const size_t off = (((size_t)b * COUT + ng) * OUTHW + u) * OUTHW + (v0 + wm + ml);
        *(float4*)(out + off) = make_float4(x0, x1, x2, x3);  // 64B contiguous per 4 lanes
    }
}

// ---------------------------------------------------------------------------
extern "C" void kernel_launch(void* const* d_in, const int* in_sizes, int n_in,
                              void* d_out, int out_size, void* d_ws, size_t ws_size,
                              hipStream_t stream) {
    const float* x    = (const float*)d_in[0];   // [16,256,128,128]
    const float* w    = (const float*)d_in[1];   // [3,3,256,512]
    const float* bias = (const float*)d_in[2];   // [512]
    float* out = (float*)d_out;                  // [16,512,64,64]

    __bf16* xb  = (__bf16*)d_ws;                 // 142.7 MB
    __bf16* bsw = xb + XB_ELEMS;                 // +9.4 MB (32B aligned)

    pad_convert <<<dim3(5, HP, NB),            256, 0, stream>>>(x, xb);
    weight_xform<<<dim3((int)(BSW_ELEMS/256)), 256, 0, stream>>>(w, bsw);
    conv_wmma   <<<dim3(2048, 4),              128, 0, stream>>>(xb, bsw, bias, out);
}
